// AdaMus_36670430773263
// MI455X (gfx1250) — matrix operand order repaired and checked by reading, not compiled, not run on hardware
//
#include <hip/hip_runtime.h>
#include <math.h>

typedef __bf16 bf16_t;
typedef __attribute__((ext_vector_type(16))) __bf16 v16bf;
typedef __attribute__((ext_vector_type(8)))  __bf16 v8bf;
typedef __attribute__((ext_vector_type(8)))  float  v8f;

union FragBF { v16bf v; v8bf h[2]; };

#define BATCH      4096
#define COM_DIM    512
#define BN_EPS_F   1e-5f
#define LAMBDA1_F  0.01f
#define MARGIN_F   1.0f

// ---------------- workspace layout ----------------
// f32 region (float offsets)
#define F32_H      0            // 4096*2048
#define F32_Z      8388608      // 4096*512
#define F32_SQ     10485760     // 4096
#define F32_STAT   10489856     // 3*5120
#define F32_SCAL   10505216     // [0]=cons, [1]=gamma
#define F32_END    10505232     // bf16 region starts here (16B aligned)
// bf16 region (bf16 element offsets from bf16 base)
#define B_X0       0            // 4096*5024
#define B_X1       20578304     // 4096*2016
#define B_X2       28835840     // 4096*1024
#define B_WRD      33030144     // 1024*5024   (W_rd^T, padded)
#define B_W1T0     38174720     // 2048*1024
#define B_W1T1     40271872     // 2048*2016
#define B_W1T2     44400640     // 2048*1024
#define B_W2T0     46497792     // 512*2048
#define B_W2T1     47546368
#define B_W2T2     48594944
#define B_V0       49643520     // 4096*1024
#define B_HB       53837824     // 4096*2048
#define B_FB       62226432     // 4096*512

// async copy of one 16B chunk: global (saddr+voffset) -> LDS[ldsaddr]
#define ASYNC_B128(ldsaddr, voff, sbase)                                   \
    asm volatile("global_load_async_to_lds_b128 %0, %1, %2"                \
                 :: "v"(ldsaddr), "v"(voff), "s"(sbase) : "memory")
#define WAIT_ASYNC(n) asm volatile("s_wait_asynccnt " #n ::: "memory")

// =====================================================================
__global__ void zero_kernel(float* __restrict__ p, int n) {
    int i = blockIdx.x * blockDim.x + threadIdx.x;
    if (i < n) p[i] = 0.0f;
}

// pad+convert activations: out[r*Kp+k] = k<K ? bf16(in[r*K+k]) : 0
__global__ void conv_pad_kernel(const float* __restrict__ in, bf16_t* __restrict__ out,
                                int K, int Kp, int total)
{
    int idx = blockIdx.x * blockDim.x + threadIdx.x;
    if (idx >= total) return;
    int r = idx / Kp, k = idx - r * Kp;
    float v = (k < K) ? in[(size_t)r * K + k] : 0.0f;
    out[idx] = (bf16_t)v;
}

// transpose+pad+convert weights: Wt[n*Kp+k] = k<K ? bf16(W[k*N+n]) : 0
__global__ void conv_transpose_kernel(const float* __restrict__ W, bf16_t* __restrict__ Wt,
                                      int K, int N, int Kp, int total)
{
    int idx = blockIdx.x * blockDim.x + threadIdx.x;
    if (idx >= total) return;
    int n = idx / Kp, k = idx - n * Kp;
    float v = (k < K) ? W[(size_t)k * N + n] : 0.0f;
    Wt[idx] = (bf16_t)v;
}

// =====================================================================
// bf16 GEMM with async double-buffered LDS staging.
// A[M,Kp] row-major bf16, Bt[N,Kp] row-major bf16 (i.e. B transposed).
// C = A @ Bt^T + bias.  OUTMODE 0: f32 out.  OUTMODE 1: bf16 out + ReLU.
// 128x128 block tile, BK=32, 8 waves (2x4), wave tile 64x32.
// =====================================================================
template<int OUTMODE>
__global__ __launch_bounds__(256)
void gemm_async_wmma(const bf16_t* __restrict__ A, const bf16_t* __restrict__ Bt,
                     const float* __restrict__ bias, void* __restrict__ Cout,
                     int N, int Kp)
{
    constexpr int LDT = 40;  // bf16 per LDS row (32 data + 8 pad); 80B stride, 16B chunks
    __shared__ __align__(16) bf16_t As[2][128 * LDT];
    __shared__ __align__(16) bf16_t Bs[2][128 * LDT];

    const int tid   = threadIdx.x;
    const int lane  = tid & 31;
    const int wid   = tid >> 5;
    const int wm    = wid >> 2;
    const int wn    = wid & 3;
    const int lhalf = lane >> 4;
    const int l15   = lane & 15;

    const int rowBase = blockIdx.y * 128;
    const int colBase = blockIdx.x * 128;

    const uint32_t ldsA[2] = { (uint32_t)(size_t)&As[0][0], (uint32_t)(size_t)&As[1][0] };
    const uint32_t ldsB[2] = { (uint32_t)(size_t)&Bs[0][0], (uint32_t)(size_t)&Bs[1][0] };
    const unsigned long long aBase = (unsigned long long)A;
    const unsigned long long bBase = (unsigned long long)Bt;

    // per-thread chunks: c and c+256 of 512 16B chunks per tile
    const int c0r = tid >> 2,         c0q = tid & 3;
    const int c1r = (tid + 256) >> 2, c1q = tid & 3;

    v8f acc[4][2];
    const v8f vz = {0.f,0.f,0.f,0.f,0.f,0.f,0.f,0.f};
#pragma unroll
    for (int mi = 0; mi < 4; ++mi)
#pragma unroll
        for (int ni = 0; ni < 2; ++ni) acc[mi][ni] = vz;

    const int nk = Kp >> 5;

    // ---- prologue: stage tile 0 into buffer 0 ----
    {
        const uint32_t la0 = ldsA[0] + (uint32_t)(c0r * 80 + c0q * 16);
        const uint32_t la1 = ldsA[0] + (uint32_t)(c1r * 80 + c1q * 16);
        const uint32_t lb0 = ldsB[0] + (uint32_t)(c0r * 80 + c0q * 16);
        const uint32_t lb1 = ldsB[0] + (uint32_t)(c1r * 80 + c1q * 16);
        ASYNC_B128(la0, (uint32_t)(((size_t)(rowBase + c0r) * Kp + c0q * 8) * 2), aBase);
        ASYNC_B128(la1, (uint32_t)(((size_t)(rowBase + c1r) * Kp + c1q * 8) * 2), aBase);
        ASYNC_B128(lb0, (uint32_t)(((size_t)(colBase + c0r) * Kp + c0q * 8) * 2), bBase);
        ASYNC_B128(lb1, (uint32_t)(((size_t)(colBase + c1r) * Kp + c1q * 8) * 2), bBase);
    }

    for (int kt = 0; kt < nk; ++kt) {
        const int buf = kt & 1;
        if (kt + 1 < nk) {
            const int nb = buf ^ 1;
            const int kB = (kt + 1) << 5;
            const uint32_t la0 = ldsA[nb] + (uint32_t)(c0r * 80 + c0q * 16);
            const uint32_t la1 = ldsA[nb] + (uint32_t)(c1r * 80 + c1q * 16);
            const uint32_t lb0 = ldsB[nb] + (uint32_t)(c0r * 80 + c0q * 16);
            const uint32_t lb1 = ldsB[nb] + (uint32_t)(c1r * 80 + c1q * 16);
            ASYNC_B128(la0, (uint32_t)(((size_t)(rowBase + c0r) * Kp + kB + c0q * 8) * 2), aBase);
            ASYNC_B128(la1, (uint32_t)(((size_t)(rowBase + c1r) * Kp + kB + c1q * 8) * 2), aBase);
            ASYNC_B128(lb0, (uint32_t)(((size_t)(colBase + c0r) * Kp + kB + c0q * 8) * 2), bBase);
            ASYNC_B128(lb1, (uint32_t)(((size_t)(colBase + c1r) * Kp + kB + c1q * 8) * 2), bBase);
            WAIT_ASYNC(0x4);   // current tile's 4 ops done (in-order), next 4 in flight
        } else {
            WAIT_ASYNC(0x0);
        }
        __syncthreads();

        v16bf afrag[4], bfrag[2];
#pragma unroll
        for (int mi = 0; mi < 4; ++mi) {
            const int ar = wm * 64 + mi * 16 + l15;
            FragBF f;
            f.h[0] = *(const v8bf*)(&As[buf][ar * LDT + lhalf * 8]);
            f.h[1] = *(const v8bf*)(&As[buf][ar * LDT + 16 + lhalf * 8]);
            afrag[mi] = f.v;
        }
#pragma unroll
        for (int ni = 0; ni < 2; ++ni) {
            const int br = wn * 32 + ni * 16 + l15;
            FragBF f;
            f.h[0] = *(const v8bf*)(&Bs[buf][br * LDT + lhalf * 8]);
            f.h[1] = *(const v8bf*)(&Bs[buf][br * LDT + 16 + lhalf * 8]);
            bfrag[ni] = f.v;
        }
#pragma unroll
        for (int mi = 0; mi < 4; ++mi)
#pragma unroll
            for (int ni = 0; ni < 2; ++ni)
                acc[mi][ni] = __builtin_amdgcn_wmma_f32_16x16x32_bf16(
                    false, afrag[mi], false, bfrag[ni],
                    (short)0, acc[mi][ni], false, false);
        __syncthreads();  // all reads of buf done before it is re-staged
    }

    // ---- epilogue ----
#pragma unroll
    for (int mi = 0; mi < 4; ++mi)
#pragma unroll
        for (int ni = 0; ni < 2; ++ni) {
            const int col = colBase + wn * 32 + ni * 16 + l15;
            const float bv = bias[col];
#pragma unroll
            for (int r = 0; r < 8; ++r) {
                const int row = rowBase + wm * 64 + mi * 16 + lhalf * 8 + r;
                float v = acc[mi][ni][r] + bv;
                if (OUTMODE == 1) {
                    ((bf16_t*)Cout)[(size_t)row * N + col] = (bf16_t)fmaxf(v, 0.0f);
                } else {
                    ((float*)Cout)[(size_t)row * N + col] = v;
                }
            }
        }
}

// =====================================================================
__global__ void bn_stats_kernel(const float* __restrict__ x,
                                float* __restrict__ fsum, float* __restrict__ fsumsq,
                                int N)
{
    const int col = blockIdx.x * 256 + threadIdx.x;
    const int r0  = blockIdx.y * (BATCH / 16);
    float s = 0.f, ss = 0.f;
    for (int r = r0; r < r0 + (BATCH / 16); ++r) {
        const float v = x[(size_t)r * N + col];
        s += v; ss += v * v;
    }
    atomicAdd(&fsum[col], s);
    atomicAdd(&fsumsq[col], ss);
}

// BN+ReLU -> bf16 (feeds next GEMM's A operand)
__global__ void bn_apply_bf16_kernel(const float* __restrict__ x,
                                     const float* __restrict__ g, const float* __restrict__ b,
                                     const float* __restrict__ fsum, const float* __restrict__ fsumsq,
                                     bf16_t* __restrict__ out, int N)
{
    const int idx = blockIdx.x * blockDim.x + threadIdx.x;
    if (idx >= BATCH * N) return;
    const int col = idx % N;
    const float invB = 1.0f / (float)BATCH;
    const float mean = fsum[col] * invB;
    const float var  = fsumsq[col] * invB - mean * mean;
    const float y = g[col] * (x[idx] - mean) * rsqrtf(var + BN_EPS_F) + b[col];
    out[idx] = (bf16_t)fmaxf(y, 0.0f);
}

// BN+ReLU, fusion += z/3 (f32)
__global__ void bn_apply_accum_kernel(const float* __restrict__ x,
                                      const float* __restrict__ g, const float* __restrict__ b,
                                      const float* __restrict__ fsum, const float* __restrict__ fsumsq,
                                      float* __restrict__ out, int N)
{
    const int idx = blockIdx.x * blockDim.x + threadIdx.x;
    if (idx >= BATCH * N) return;
    const int col = idx % N;
    const float invB = 1.0f / (float)BATCH;
    const float mean = fsum[col] * invB;
    const float var  = fsumsq[col] * invB - mean * mean;
    const float y = g[col] * (x[idx] - mean) * rsqrtf(var + BN_EPS_F) + b[col];
    out[idx] += fmaxf(y, 0.0f) * (1.0f / 3.0f);
}

// sq[row] = ||fusion_row||^2, and emit bf16 fusion for the Gram kernel
__global__ void row_sumsq_kernel(const float* __restrict__ f, float* __restrict__ sq,
                                 bf16_t* __restrict__ fb)
{
    __shared__ float red[128];
    const int row = blockIdx.x;
    const int tid = threadIdx.x;
    float s = 0.f;
#pragma unroll
    for (int j = tid; j < COM_DIM; j += 128) {
        const float v = f[(size_t)row * COM_DIM + j];
        fb[(size_t)row * COM_DIM + j] = (bf16_t)v;
        s += v * v;
    }
    red[tid] = s; __syncthreads();
    for (int st = 64; st > 0; st >>= 1) {
        if (tid < st) red[tid] += red[tid + st];
        __syncthreads();
    }
    if (tid == 0) sq[row] = red[0];
}

__global__ void gamma_l1_kernel(const float* __restrict__ g0, const float* __restrict__ g1,
                                const float* __restrict__ g2, float* __restrict__ outg)
{
    __shared__ float red[256];
    const int tid = threadIdx.x;
    float s = 0.f;
    for (int i = tid; i < COM_DIM; i += 256)
        s += fabsf(g0[i]) + fabsf(g1[i]) + fabsf(g2[i]);
    red[tid] = s; __syncthreads();
    for (int st = 128; st > 0; st >>= 1) {
        if (tid < st) red[tid] += red[tid + st];
        __syncthreads();
    }
    if (tid == 0) *outg = red[0] / (float)COM_DIM;
}

// =====================================================================
// Gram + contrastive loss, async-staged bf16 WMMA, Kp = 512.
// =====================================================================
__global__ __launch_bounds__(256)
void gram_loss_wmma(const bf16_t* __restrict__ Fb, const float* __restrict__ S,
                    const float* __restrict__ sq, float* __restrict__ cons)
{
    constexpr int LDT = 40;
    constexpr int KP  = COM_DIM;
    __shared__ __align__(16) bf16_t As[2][128 * LDT];
    __shared__ __align__(16) bf16_t Bs[2][128 * LDT];
    __shared__ float red[256];

    const int tid   = threadIdx.x;
    const int lane  = tid & 31;
    const int wid   = tid >> 5;
    const int wm    = wid >> 2;
    const int wn    = wid & 3;
    const int lhalf = lane >> 4;
    const int l15   = lane & 15;

    const int rowBase = blockIdx.y * 128;
    const int colBase = blockIdx.x * 128;

    const uint32_t ldsA[2] = { (uint32_t)(size_t)&As[0][0], (uint32_t)(size_t)&As[1][0] };
    const uint32_t ldsB[2] = { (uint32_t)(size_t)&Bs[0][0], (uint32_t)(size_t)&Bs[1][0] };
    const unsigned long long fBase = (unsigned long long)Fb;

    const int c0r = tid >> 2,         c0q = tid & 3;
    const int c1r = (tid + 256) >> 2, c1q = tid & 3;

    v8f acc[4][2];
    const v8f vz = {0.f,0.f,0.f,0.f,0.f,0.f,0.f,0.f};
#pragma unroll
    for (int mi = 0; mi < 4; ++mi)
#pragma unroll
        for (int ni = 0; ni < 2; ++ni) acc[mi][ni] = vz;

    {
        const uint32_t off = (uint32_t)(c0r * 80 + c0q * 16);
        const uint32_t off1 = (uint32_t)(c1r * 80 + c1q * 16);
        ASYNC_B128(ldsA[0] + off,  (uint32_t)(((size_t)(rowBase + c0r) * KP + c0q * 8) * 2), fBase);
        ASYNC_B128(ldsA[0] + off1, (uint32_t)(((size_t)(rowBase + c1r) * KP + c1q * 8) * 2), fBase);
        ASYNC_B128(ldsB[0] + off,  (uint32_t)(((size_t)(colBase + c0r) * KP + c0q * 8) * 2), fBase);
        ASYNC_B128(ldsB[0] + off1, (uint32_t)(((size_t)(colBase + c1r) * KP + c1q * 8) * 2), fBase);
    }

    const int nk = KP >> 5;
    for (int kt = 0; kt < nk; ++kt) {
        const int buf = kt & 1;
        if (kt + 1 < nk) {
            const int nb = buf ^ 1;
            const int kB = (kt + 1) << 5;
            const uint32_t off  = (uint32_t)(c0r * 80 + c0q * 16);
            const uint32_t off1 = (uint32_t)(c1r * 80 + c1q * 16);
            ASYNC_B128(ldsA[nb] + off,  (uint32_t)(((size_t)(rowBase + c0r) * KP + kB + c0q * 8) * 2), fBase);
            ASYNC_B128(ldsA[nb] + off1, (uint32_t)(((size_t)(rowBase + c1r) * KP + kB + c1q * 8) * 2), fBase);
            ASYNC_B128(ldsB[nb] + off,  (uint32_t)(((size_t)(colBase + c0r) * KP + kB + c0q * 8) * 2), fBase);
            ASYNC_B128(ldsB[nb] + off1, (uint32_t)(((size_t)(colBase + c1r) * KP + kB + c1q * 8) * 2), fBase);
            WAIT_ASYNC(0x4);
        } else {
            WAIT_ASYNC(0x0);
        }
        __syncthreads();

        v16bf afrag[4], bfrag[2];
#pragma unroll
        for (int mi = 0; mi < 4; ++mi) {
            const int ar = wm * 64 + mi * 16 + l15;
            FragBF f;
            f.h[0] = *(const v8bf*)(&As[buf][ar * LDT + lhalf * 8]);
            f.h[1] = *(const v8bf*)(&As[buf][ar * LDT + 16 + lhalf * 8]);
            afrag[mi] = f.v;
        }
#pragma unroll
        for (int ni = 0; ni < 2; ++ni) {
            const int br = wn * 32 + ni * 16 + l15;
            FragBF f;
            f.h[0] = *(const v8bf*)(&Bs[buf][br * LDT + lhalf * 8]);
            f.h[1] = *(const v8bf*)(&Bs[buf][br * LDT + 16 + lhalf * 8]);
            bfrag[ni] = f.v;
        }
#pragma unroll
        for (int mi = 0; mi < 4; ++mi)
#pragma unroll
            for (int ni = 0; ni < 2; ++ni)
                acc[mi][ni] = __builtin_amdgcn_wmma_f32_16x16x32_bf16(
                    false, afrag[mi], false, bfrag[ni],
                    (short)0, acc[mi][ni], false, false);
        __syncthreads();
    }

    // ---- loss epilogue ----
    const float invB2 = 1.0f / ((float)BATCH * (float)BATCH);
    float local = 0.f;
#pragma unroll
    for (int mi = 0; mi < 4; ++mi)
#pragma unroll
        for (int ni = 0; ni < 2; ++ni) {
            const int j   = colBase + wn * 32 + ni * 16 + l15;
            const float sqj = sq[j];
#pragma unroll
            for (int r = 0; r < 8; ++r) {
                const int i = rowBase + wm * 64 + mi * 16 + lhalf * 8 + r;
                if (i == j) continue;
                const float dist = (sq[i] + sqj - 2.0f * acc[mi][ni][r]) * invB2;
                const bool a = (S[(size_t)i * BATCH + j] > 0.6f) ||
                               (S[(size_t)j * BATCH + i] > 0.6f);
                if (a) {
                    local += dist;
                } else {
                    const float t = MARGIN_F - sqrtf(fmaxf(dist, 0.0f));
                    local += (t > 0.0f) ? t * t : 0.0f;
                }
            }
        }
    red[tid] = local; __syncthreads();
    for (int st = 128; st > 0; st >>= 1) {
        if (tid < st) red[tid] += red[tid + st];
        __syncthreads();
    }
    if (tid == 0) atomicAdd(cons, red[0]);
}

__global__ void finalize_kernel(const float* __restrict__ scal, float* __restrict__ loss)
{
    loss[0] = scal[0] * 0.5f + LAMBDA1_F * scal[1];
}

// =====================================================================
extern "C" void kernel_launch(void* const* d_in, const int* in_sizes, int n_in,
                              void* d_out, int out_size, void* d_ws, size_t ws_size,
                              hipStream_t stream)
{
    (void)in_sizes; (void)n_in; (void)out_size; (void)ws_size;

    const float* x0  = (const float*)d_in[0];
    const float* x1  = (const float*)d_in[1];
    const float* x2  = (const float*)d_in[2];
    const float* S   = (const float*)d_in[3];
    const float* Wrd = (const float*)d_in[5];
    const float* brd = (const float*)d_in[6];
    const float* encp[3][8];
    for (int v = 0; v < 3; ++v)
        for (int k = 0; k < 8; ++k)
            encp[v][k] = (const float*)d_in[7 + v * 8 + k];

    float*  ws     = (float*)d_ws;
    bf16_t* bw     = (bf16_t*)(ws + F32_END);
    float*  out    = (float*)d_out;
    float*  fusion = out;
    float*  lossp  = out + (size_t)BATCH * COM_DIM;

    float* h  = ws + F32_H;
    float* z  = ws + F32_Z;
    float* sq = ws + F32_SQ;
    float* sc = ws + F32_SCAL;

    bf16_t* x0b  = bw + B_X0;
    bf16_t* x1b  = bw + B_X1;
    bf16_t* x2b  = bw + B_X2;
    bf16_t* wrdt = bw + B_WRD;
    bf16_t* w1t[3] = { bw + B_W1T0, bw + B_W1T1, bw + B_W1T2 };
    bf16_t* w2t[3] = { bw + B_W2T0, bw + B_W2T1, bw + B_W2T2 };
    bf16_t* v0b  = bw + B_V0;
    bf16_t* hb   = bw + B_HB;
    bf16_t* fb   = bw + B_FB;

    // ---- zero fusion+loss and stats/scalars ----
    {
        const int n1 = BATCH * COM_DIM + 1;
        zero_kernel<<<(n1 + 255) / 256, 256, 0, stream>>>(out, n1);
        const int n2 = 3 * 5120 + 8;
        zero_kernel<<<(n2 + 255) / 256, 256, 0, stream>>>(ws + F32_STAT, n2);
    }

    // ---- one-time bf16 conversions (pad K to mult of 32; transpose weights) ----
    {
        int t;
        t = BATCH * 5024; conv_pad_kernel<<<(t + 255) / 256, 256, 0, stream>>>(x0, x0b, 5000, 5024, t);
        t = BATCH * 2016; conv_pad_kernel<<<(t + 255) / 256, 256, 0, stream>>>(x1, x1b, 2000, 2016, t);
        t = BATCH * 1024; conv_pad_kernel<<<(t + 255) / 256, 256, 0, stream>>>(x2, x2b, 1000, 1024, t);
        t = 1024 * 5024;  conv_transpose_kernel<<<(t + 255) / 256, 256, 0, stream>>>(Wrd, wrdt, 5000, 1024, 5024, t);
        t = 2048 * 1024;  conv_transpose_kernel<<<(t + 255) / 256, 256, 0, stream>>>(encp[0][0], w1t[0], 1024, 2048, 1024, t);
        t = 2048 * 2016;  conv_transpose_kernel<<<(t + 255) / 256, 256, 0, stream>>>(encp[1][0], w1t[1], 2000, 2048, 2016, t);
        t = 2048 * 1024;  conv_transpose_kernel<<<(t + 255) / 256, 256, 0, stream>>>(encp[2][0], w1t[2], 1000, 2048, 1024, t);
        for (int v = 0; v < 3; ++v) {
            t = 512 * 2048;
            conv_transpose_kernel<<<(t + 255) / 256, 256, 0, stream>>>(encp[v][4], w2t[v], 2048, 512, 2048, t);
        }
    }

    // ---- rd_net: v0b = relu(x0 @ W_rd + b_rd) -> bf16 directly ----
    gemm_async_wmma<1><<<dim3(1024 / 128, BATCH / 128), 256, 0, stream>>>(
        x0b, wrdt, brd, v0b, 1024, 5024);

    const bf16_t* ain[3] = { v0b, x1b, x2b };
    const int     aKp[3] = { 1024, 2016, 1024 };

    for (int v = 0; v < 3; ++v) {
        float* st   = ws + F32_STAT + v * 5120;
        float *sum1 = st, *ssq1 = st + 2048, *sum2 = st + 4096, *ssq2 = st + 4608;
        const float *b1 = encp[v][1], *g1 = encp[v][2], *be1 = encp[v][3];
        const float *b2 = encp[v][5], *g2 = encp[v][6], *be2 = encp[v][7];

        gemm_async_wmma<0><<<dim3(2048 / 128, BATCH / 128), 256, 0, stream>>>(
            ain[v], w1t[v], b1, h, 2048, aKp[v]);
        bn_stats_kernel<<<dim3(2048 / 256, 16), 256, 0, stream>>>(h, sum1, ssq1, 2048);
        bn_apply_bf16_kernel<<<(BATCH * 2048) / 256, 256, 0, stream>>>(
            h, g1, be1, sum1, ssq1, hb, 2048);
        gemm_async_wmma<0><<<dim3(COM_DIM / 128, BATCH / 128), 256, 0, stream>>>(
            hb, w2t[v], b2, z, COM_DIM, 2048);
        bn_stats_kernel<<<dim3(COM_DIM / 256, 16), 256, 0, stream>>>(z, sum2, ssq2, COM_DIM);
        bn_apply_accum_kernel<<<(BATCH * COM_DIM) / 256, 256, 0, stream>>>(
            z, g2, be2, sum2, ssq2, fusion, COM_DIM);
    }

    row_sumsq_kernel<<<BATCH, 128, 0, stream>>>(fusion, sq, fb);
    gamma_l1_kernel<<<1, 256, 0, stream>>>(encp[0][6], encp[1][6], encp[2][6], &sc[1]);
    gram_loss_wmma<<<dim3(BATCH / 128, BATCH / 128), 256, 0, stream>>>(fb, S, sq, &sc[0]);
    finalize_kernel<<<1, 1, 0, stream>>>(sc, lossp);
}
